// HyPConv_8881992368544
// MI455X (gfx1250) — matrix operand order repaired and checked by reading, not compiled, not run on hardware
//
#include <hip/hip_runtime.h>
#include <hip/hip_bf16.h>

// Problem constants (match reference)
static constexpr int BB = 8;      // batch
static constexpr int NN = 4096;   // nodes
static constexpr int EE = 2048;   // hyperedges
static constexpr int CC = 128;    // C1 == C2 == 128

// WMMA tiling constants
static constexpr int KSTEP = 32;                  // K per wmma step
static constexpr int KPAIRS = KSTEP / 2;          // 16 packed uints per column
static constexpr int PANEL_UINTS = CC * KPAIRS;   // 2048 uints per packed B panel (global)
static constexpr int BP_PITCH = 20;               // LDS col pitch in uints (80B, bank-friendly)
static constexpr int A_PITCH = 40;                // LDS A row pitch in shorts (80B, bank-friendly)

typedef __attribute__((ext_vector_type(16))) __bf16 v16bf;
typedef __attribute__((ext_vector_type(2)))  __bf16 v2bf;
typedef __attribute__((ext_vector_type(8)))  float  v8f;
typedef __attribute__((ext_vector_type(4)))  int    v4i;

#if __has_builtin(__builtin_amdgcn_global_load_async_to_lds_b128) && \
    __has_builtin(__builtin_amdgcn_s_wait_asynccnt)
#define HAS_ASYNC_LDS 1
#else
#define HAS_ASYNC_LDS 0
#endif

union AFrag { v16bf v; unsigned int u[8]; };

// Packed 2xf32 -> 2xbf16 (low = first arg).
__device__ __forceinline__ unsigned int pack_bf16(float lo, float hi) {
#if __has_builtin(__builtin_amdgcn_cvt_pk_bf16_f32)
    union { v2bf h; unsigned int u; } cv;
    cv.h = __builtin_amdgcn_cvt_pk_bf16_f32(lo, hi);
    return cv.u;
#else
    // Native fptrunc: backend legalizes to v_cvt_pk_bf16_f32 where available.
    union { __bf16 h[2]; unsigned int u; } cv;
    cv.h[0] = (__bf16)lo;
    cv.h[1] = (__bf16)hi;
    return cv.u;
#endif
}

// ---------------------------------------------------------------------------
// WMMA tile core. Preloads A (2x b128) and all 8 B fragments (16x b128),
// then issues 8 back-to-back WMMAs (disjoint D, shared A -> no hazard NOPs).
// ---------------------------------------------------------------------------
__device__ __forceinline__ void wmma_tile_compute(const unsigned short* Asub,
                                                  const unsigned int* BsubP,
                                                  v8f* acc, int wave, int lane) {
    const int half = lane >> 4;
    const int rowA = (wave << 4) + (lane & 15);

    const char* abase = (const char*)Asub + rowA * (A_PITCH * 2) + 32 * half;
    uint4 a0 = *(const uint4*)(abase);
    uint4 a1 = *(const uint4*)(abase + 16);
    AFrag a;
    a.u[0] = a0.x; a.u[1] = a0.y; a.u[2] = a0.z; a.u[3] = a0.w;
    a.u[4] = a1.x; a.u[5] = a1.y; a.u[6] = a1.z; a.u[7] = a1.w;

    const int col = lane & 15;
    AFrag bf[8];
#pragma unroll
    for (int t = 0; t < 8; ++t) {
        const char* bbase = (const char*)(BsubP + (t * 16 + col) * BP_PITCH) + 32 * half;
        uint4 b0 = *(const uint4*)(bbase);
        uint4 b1 = *(const uint4*)(bbase + 16);
        bf[t].u[0] = b0.x; bf[t].u[1] = b0.y; bf[t].u[2] = b0.z; bf[t].u[3] = b0.w;
        bf[t].u[4] = b1.x; bf[t].u[5] = b1.y; bf[t].u[6] = b1.z; bf[t].u[7] = b1.w;
    }
#pragma unroll
    for (int t = 0; t < 8; ++t)
        acc[t] = __builtin_amdgcn_wmma_f32_16x16x32_bf16(
            false, a.v, false, bf[t].v, (short)0, acc[t], false, false);
}

// ---------------- staging helpers ------------------------------------------
struct AReg { float4 f[4]; };
struct BReg { unsigned int u[8]; };

#if HAS_ASYNC_LDS
// Async DMA: one 16B chunk = 4 k-pairs of one column -> stays inside padded
// column. 512 chunks, 2 per thread. Tracked by ASYNCcnt.
__device__ __forceinline__ void async_stage_b(const unsigned int* __restrict__ src,
                                              unsigned int* BsubP, int tid) {
#pragma unroll
    for (int i = 0; i < 2; ++i) {
        int chunk = tid + i * 256;               // 0..511
        int c = chunk >> 2;                      // column
        int kp4 = (chunk & 3) * 4;               // k-pair group
        __builtin_amdgcn_global_load_async_to_lds_b128(
            (__attribute__((address_space(1))) v4i*)(src + (long)chunk * 4),
            (__attribute__((address_space(3))) v4i*)(BsubP + c * BP_PITCH + kp4),
            0, 0);
    }
}
__device__ __forceinline__ void async_wait() {
    __builtin_amdgcn_s_wait_asynccnt(0);
}
#endif

__device__ __forceinline__ void load_b_panel(const unsigned int* __restrict__ src,
                                             int tid, BReg& br) {
#pragma unroll
    for (int i = 0; i < 8; ++i) br.u[i] = src[tid + i * 256];  // contiguous, coalesced
}

__device__ __forceinline__ void store_b_panel(unsigned int* BsubP, int tid,
                                              const BReg& br) {
#pragma unroll
    for (int i = 0; i < 8; ++i) {
        int idx = tid + i * 256;
        BsubP[(idx >> 4) * BP_PITCH + (idx & 15)] = br.u[i];
    }
}

// A tile from row-major fp32 (stride ld): rows 0..127, K cols k0..k0+31
__device__ __forceinline__ void load_a_rowmajor(const float* __restrict__ base,
                                                long ld, int k0, int tid, AReg& ar) {
#pragma unroll
    for (int i = 0; i < 4; ++i) {
        int row = (tid >> 3) + i * 32;
        int c = (tid & 7) * 4;
        ar.f[i] = *(const float4*)(base + (long)row * ld + k0 + c);
    }
}

__device__ __forceinline__ void store_a_rowmajor(unsigned short* Asub, int tid,
                                                 const AReg& ar) {
#pragma unroll
    for (int i = 0; i < 4; ++i) {
        int row = (tid >> 3) + i * 32;
        int c = (tid & 7) * 4;
        uint2 p;
        p.x = pack_bf16(ar.f[i].x, ar.f[i].y);
        p.y = pack_bf16(ar.f[i].z, ar.f[i].w);
        *(uint2*)(Asub + row * A_PITCH + c) = p;   // single ds_store_b64
    }
}

// A tile = H^T block: read H[k0+n][e0+c] (coalesced along E), store transposed
__device__ __forceinline__ void load_a_transpose(const float* __restrict__ Hb,
                                                 long e0, int k0, int tid, AReg& ar) {
#pragma unroll
    for (int i = 0; i < 4; ++i) {
        int n = (tid >> 5) + i * 8;
        int c = (tid & 31) * 4;
        ar.f[i] = *(const float4*)(Hb + (long)(k0 + n) * EE + e0 + c);
    }
}

__device__ __forceinline__ void store_a_transpose(unsigned short* Asub, int tid,
                                                  const AReg& ar) {
#pragma unroll
    for (int i = 0; i < 4; ++i) {
        int n = (tid >> 5) + i * 8;
        int c = (tid & 31) * 4;
        unsigned int u01 = pack_bf16(ar.f[i].x, ar.f[i].y);
        unsigned int u23 = pack_bf16(ar.f[i].z, ar.f[i].w);
        Asub[(c + 0) * A_PITCH + n] = (unsigned short)(u01);
        Asub[(c + 1) * A_PITCH + n] = (unsigned short)(u01 >> 16);
        Asub[(c + 2) * A_PITCH + n] = (unsigned short)(u23);
        Asub[(c + 3) * A_PITCH + n] = (unsigned short)(u23 >> 16);
    }
}

// ---------------------------------------------------------------------------
// Degree kernel: one pass over H -> deg_v (wave reduce) and deg_e (global
// float atomics; summands are exactly 0/1 -> deterministic & exact).
// ---------------------------------------------------------------------------
__global__ __launch_bounds__(256) void degree_kernel(const float* __restrict__ H,
                                                     float* __restrict__ deg_e,
                                                     float* __restrict__ deg_v) {
    __shared__ float sv[64];
    const int tid = threadIdx.x;
    const int b = blockIdx.y;
    const int n0 = blockIdx.x * 64;
    if (tid < 64) sv[tid] = 0.f;
    __syncthreads();

    const float* Hb = H + ((long)b * NN + n0) * (long)EE;
    float colAcc[8];
#pragma unroll
    for (int j = 0; j < 8; ++j) colAcc[j] = 0.f;

    for (int r = 0; r < 64; ++r) {
        const float* row = Hb + (long)r * EE;
        float rs = 0.f;
#pragma unroll
        for (int j = 0; j < 8; ++j) {
            float h = row[tid + 256 * j];
            rs += h;
            colAcc[j] += h;
        }
#pragma unroll
        for (int off = 16; off > 0; off >>= 1) rs += __shfl_down(rs, off);
        if ((tid & 31) == 0) atomicAdd(&sv[r], rs);
    }
    __syncthreads();
    if (tid < 64) deg_v[(long)b * NN + n0 + tid] = sv[tid];
#pragma unroll
    for (int j = 0; j < 8; ++j)
        atomicAdd(&deg_e[(long)b * EE + tid + 256 * j], colAcc[j]);
}

__global__ void recip_kernel(float* __restrict__ a, int n) {
    int i = blockIdx.x * blockDim.x + threadIdx.x;
    if (i < n) { float s = a[i]; a[i] = (s > 0.f) ? (1.f / s) : 0.f; }
}

// ---------------------------------------------------------------------------
// fc: x2 = x @ W + b -> packed bf16 K-pair panels. K=128 (4 steps), tiny.
// ---------------------------------------------------------------------------
__global__ __launch_bounds__(256) void fc_kernel(const float* __restrict__ x,
                                                 const float* __restrict__ W,
                                                 const float* __restrict__ bias,
                                                 unsigned int* __restrict__ x2p) {
    __shared__ __align__(16) unsigned short Asub[128 * A_PITCH];
    __shared__ __align__(16) unsigned int BsubP[128 * BP_PITCH];
    const int tid = threadIdx.x, wave = tid >> 5, lane = tid & 31;
    const long m0 = (long)blockIdx.x * 128;

    v8f acc[8];
#pragma unroll
    for (int t = 0; t < 8; ++t)
#pragma unroll
        for (int i = 0; i < 8; ++i) acc[t][i] = 0.f;

    for (int k0 = 0; k0 < CC; k0 += KSTEP) {
        AReg ar;
        load_a_rowmajor(x + m0 * CC, CC, k0, tid, ar);
        store_a_rowmajor(Asub, tid, ar);
        // W panel: fp32 -> packed K-pair columns (W is tiny, L2 resident)
#pragma unroll
        for (int i = 0; i < 8; ++i) {
            int idx = tid + i * 256;
            int c = idx >> 4;
            int kp = idx & 15;
            BsubP[c * BP_PITCH + kp] = pack_bf16(W[(k0 + 2 * kp) * (long)CC + c],
                                                 W[(k0 + 2 * kp + 1) * (long)CC + c]);
        }
        __syncthreads();
        wmma_tile_compute(Asub, BsubP, acc, wave, lane);
        __syncthreads();
    }

    const int half = lane >> 4, col = lane & 15;
#pragma unroll
    for (int t = 0; t < 8; ++t) {
        int c = t * 16 + col;
        float bb = bias[c];
#pragma unroll
        for (int r = 0; r < 8; r += 2) {
            long m = m0 + wave * 16 + 8 * half + r;   // even
            long panel = m >> 5;
            int kp = (int)((m & 31) >> 1);
            x2p[panel * PANEL_UINTS + c * KPAIRS + kp] =
                pack_bf16(acc[t][r] + bb, acc[t][r + 1] + bb);
        }
    }
}

// ---------------------------------------------------------------------------
// v2e: e = (H^T @ x2) * recip_deg_e. Double-buffered LDS; B panels staged by
// async global->LDS DMA when available. grid = (EE/128, BB).
// ---------------------------------------------------------------------------
__global__ __launch_bounds__(256) void v2e_kernel(const float* __restrict__ H,
                                                  const unsigned int* __restrict__ x2p,
                                                  const float* __restrict__ recip_e,
                                                  unsigned int* __restrict__ ep) {
    __shared__ __align__(16) unsigned short Asub[2][128 * A_PITCH];
    __shared__ __align__(16) unsigned int BsubP[2][128 * BP_PITCH];
    const int tid = threadIdx.x, wave = tid >> 5, lane = tid & 31;
    const int b = blockIdx.y;
    const long e0 = (long)blockIdx.x * 128;
    const float* Hb = H + (long)b * NN * EE;
    const unsigned int* Bpanels = x2p + (long)b * (NN / KSTEP) * PANEL_UINTS;

    v8f acc[8];
#pragma unroll
    for (int t = 0; t < 8; ++t)
#pragma unroll
        for (int i = 0; i < 8; ++i) acc[t][i] = 0.f;

    constexpr int NKT = NN / KSTEP;   // 128
    AReg ar; BReg br;
    load_a_transpose(Hb, e0, 0, tid, ar);
    store_a_transpose(Asub[0], tid, ar);
#if HAS_ASYNC_LDS
    async_stage_b(Bpanels, BsubP[0], tid);
    async_wait();
#else
    load_b_panel(Bpanels, tid, br);
    store_b_panel(BsubP[0], tid, br);
#endif
    __syncthreads();

    for (int kt = 0; kt < NKT; ++kt) {
        int cur = kt & 1, nxt = cur ^ 1;
        if (kt + 1 < NKT) {            // issue next-tile loads before compute
            load_a_transpose(Hb, e0, (kt + 1) * KSTEP, tid, ar);
#if HAS_ASYNC_LDS
            async_stage_b(Bpanels + (long)(kt + 1) * PANEL_UINTS, BsubP[nxt], tid);
#else
            load_b_panel(Bpanels + (long)(kt + 1) * PANEL_UINTS, tid, br);
#endif
        }
        wmma_tile_compute(Asub[cur], BsubP[cur], acc, wave, lane);
        if (kt + 1 < NKT) {
            store_a_transpose(Asub[nxt], tid, ar);
#if !HAS_ASYNC_LDS
            store_b_panel(BsubP[nxt], tid, br);
#endif
        }
#if HAS_ASYNC_LDS
        async_wait();
#endif
        __syncthreads();
    }

    const int half = lane >> 4, col = lane & 15;
#pragma unroll
    for (int t = 0; t < 8; ++t) {
        int c = t * 16 + col;
#pragma unroll
        for (int r = 0; r < 8; r += 2) {
            long row = e0 + wave * 16 + 8 * half + r;     // even
            float s0 = recip_e[(long)b * EE + row];
            float s1 = recip_e[(long)b * EE + row + 1];
            long panel = (long)b * (EE / KSTEP) + (row >> 5);
            int kp = (int)((row & 31) >> 1);
            ep[panel * PANEL_UINTS + c * KPAIRS + kp] =
                pack_bf16(acc[t][r] * s0, acc[t][r + 1] * s1);
        }
    }
}

// ---------------------------------------------------------------------------
// e2v: v = (H @ e) * recip_deg_v, fp32 out. Double-buffered; async B staging.
// grid = (NN/128, BB).
// ---------------------------------------------------------------------------
__global__ __launch_bounds__(256) void e2v_kernel(const float* __restrict__ H,
                                                  const unsigned int* __restrict__ ep,
                                                  const float* __restrict__ recip_v,
                                                  float* __restrict__ out) {
    __shared__ __align__(16) unsigned short Asub[2][128 * A_PITCH];
    __shared__ __align__(16) unsigned int BsubP[2][128 * BP_PITCH];
    const int tid = threadIdx.x, wave = tid >> 5, lane = tid & 31;
    const int b = blockIdx.y;
    const long n0 = (long)blockIdx.x * 128;
    const float* Hrows = H + ((long)b * NN + n0) * (long)EE;
    const unsigned int* Bpanels = ep + (long)b * (EE / KSTEP) * PANEL_UINTS;

    v8f acc[8];
#pragma unroll
    for (int t = 0; t < 8; ++t)
#pragma unroll
        for (int i = 0; i < 8; ++i) acc[t][i] = 0.f;

    constexpr int NKT = EE / KSTEP;   // 64
    AReg ar; BReg br;
    load_a_rowmajor(Hrows, EE, 0, tid, ar);
    store_a_rowmajor(Asub[0], tid, ar);
#if HAS_ASYNC_LDS
    async_stage_b(Bpanels, BsubP[0], tid);
    async_wait();
#else
    load_b_panel(Bpanels, tid, br);
    store_b_panel(BsubP[0], tid, br);
#endif
    __syncthreads();

    for (int kt = 0; kt < NKT; ++kt) {
        int cur = kt & 1, nxt = cur ^ 1;
        if (kt + 1 < NKT) {
            load_a_rowmajor(Hrows, EE, (kt + 1) * KSTEP, tid, ar);
#if HAS_ASYNC_LDS
            async_stage_b(Bpanels + (long)(kt + 1) * PANEL_UINTS, BsubP[nxt], tid);
#else
            load_b_panel(Bpanels + (long)(kt + 1) * PANEL_UINTS, tid, br);
#endif
        }
        wmma_tile_compute(Asub[cur], BsubP[cur], acc, wave, lane);
        if (kt + 1 < NKT) {
            store_a_rowmajor(Asub[nxt], tid, ar);
#if !HAS_ASYNC_LDS
            store_b_panel(BsubP[nxt], tid, br);
#endif
        }
#if HAS_ASYNC_LDS
        async_wait();
#endif
        __syncthreads();
    }

    const int half = lane >> 4, col = lane & 15;
#pragma unroll
    for (int t = 0; t < 8; ++t)
#pragma unroll
        for (int r = 0; r < 8; ++r) {
            long row = n0 + wave * 16 + r + 8 * half;
            float s = recip_v[(long)b * NN + row];
            out[((long)b * NN + row) * CC + t * 16 + col] = acc[t][r] * s;
        }
}

// ---------------------------------------------------------------------------
extern "C" void kernel_launch(void* const* d_in, const int* in_sizes, int n_in,
                              void* d_out, int out_size, void* d_ws, size_t ws_size,
                              hipStream_t stream) {
    const float* x    = (const float*)d_in[0];   // [B,N,C1]
    const float* H    = (const float*)d_in[1];   // [B,N,E]
    const float* W    = (const float*)d_in[2];   // [C1,C2]
    const float* bias = (const float*)d_in[3];   // [C2]
    float* out = (float*)d_out;                  // [B,N,C2] fp32

    char* ws = (char*)d_ws;
    const size_t X2_BYTES = (size_t)(BB * NN / KSTEP) * PANEL_UINTS * 4; // 8.39 MB
    const size_t EB_BYTES = (size_t)(BB * EE / KSTEP) * PANEL_UINTS * 4; // 4.19 MB
    unsigned int* x2p = (unsigned int*)ws;
    unsigned int* ep  = (unsigned int*)(ws + X2_BYTES);
    float* deg   = (float*)(ws + X2_BYTES + EB_BYTES);
    float* deg_e = deg;                    // [B*E]
    float* deg_v = deg + (size_t)BB * EE;  // [B*N]

    (void)hipMemsetAsync(deg_e, 0, (size_t)BB * EE * sizeof(float), stream);

    degree_kernel<<<dim3(NN / 64, BB), 256, 0, stream>>>(H, deg_e, deg_v);

    int ndeg = BB * EE + BB * NN;
    recip_kernel<<<(ndeg + 255) / 256, 256, 0, stream>>>(deg, ndeg);

    fc_kernel<<<dim3((BB * NN) / 128), 256, 0, stream>>>(x, W, bias, x2p);

    v2e_kernel<<<dim3(EE / 128, BB), 256, 0, stream>>>(H, x2p, deg_e, ep);

    e2v_kernel<<<dim3(NN / 128, BB), 256, 0, stream>>>(H, ep, deg_v, out);
}
